// Attention_57543971832768
// MI455X (gfx1250) — matrix operand order repaired and checked by reading, not compiled, not run on hardware
//
#include <hip/hip_runtime.h>

// ---------------------------------------------------------------------------
// Types for CDNA5 WMMA (wave32, 16x16x32 bf16 -> f32)
// ---------------------------------------------------------------------------
typedef __bf16 bf16;
typedef __attribute__((ext_vector_type(16))) bf16  v16bf;
typedef __attribute__((ext_vector_type(8)))  bf16  v8bf;
typedef __attribute__((ext_vector_type(8)))  float v8f;
typedef __attribute__((ext_vector_type(4)))  int   v4i;

union V16U { v16bf v; v8bf h[2]; };

__device__ __forceinline__ v16bf load16(const bf16* lo, const bf16* hi) {
    V16U u;
    u.h[0] = *reinterpret_cast<const v8bf*>(lo);
    u.h[1] = *reinterpret_cast<const v8bf*>(hi);
    return u.v;
}

#define WMMA_BF16(A, B, C) \
    __builtin_amdgcn_wmma_f32_16x16x32_bf16(false, (A), false, (B), (short)0, (C), false, false)

// CDNA5 async global->LDS path (ASYNCcnt), guarded so host pass / older
// toolchains fall back to a register-staged pipeline with identical structure.
#if __has_builtin(__builtin_amdgcn_global_load_async_to_lds_b128) && \
    __has_builtin(__builtin_amdgcn_s_wait_asynccnt)
#define HAVE_ASYNC_LDS 1
#else
#define HAVE_ASYNC_LDS 0
#endif

#if HAVE_ASYNC_LDS
// Builtin signature (from compiler diagnostic): param0 = AS1 (global) v4i*,
// param1 = AS3 (LDS) v4i*, then imm offset, imm cpol.
typedef __attribute__((address_space(1))) v4i* gv4i_p;
typedef __attribute__((address_space(3))) v4i* lv4i_p;

__device__ __forceinline__ void async16(const bf16* g, bf16* l) {
    __builtin_amdgcn_global_load_async_to_lds_b128(
        (gv4i_p)(bf16*)g, (lv4i_p)l, 0, 0);
}
__device__ __forceinline__ void wait_async() {
    __builtin_amdgcn_s_wait_asynccnt(0);
}
#else
__device__ __forceinline__ void wait_async() {}
#endif

// Problem constants
#define B_    2
#define S_    2048
#define D_    2048
#define NH_   32
#define NKV_  8
#define HD_   64
#define TK_   3072   // (32 + 2*8) * 64

// ---------------------------------------------------------------------------
// 1) f32 -> bf16 cast
// ---------------------------------------------------------------------------
__global__ void cast_f32_bf16(const float* __restrict__ src,
                              bf16* __restrict__ dst, int n) {
    int i = blockIdx.x * blockDim.x + threadIdx.x;
    if (i < n) dst[i] = (bf16)src[i];
}

// ---------------------------------------------------------------------------
// 2) GEMM: C[M,N] (f32) = A[M,K] (bf16, row-major) * B[N,K]^T (bf16, row-major)
//    128x128 block tile, BK=32, 8 waves (2x4), 64x32 per wave.
//    Double-buffered LDS, async (DMA) tile fill overlapped with WMMA.
// ---------------------------------------------------------------------------
__global__ __launch_bounds__(256)
void gemm_bf16_nt(const bf16* __restrict__ A, const bf16* __restrict__ B,
                  float* __restrict__ C, int M, int N, int K) {
    constexpr int LDT = 40;  // 32 K-halves + 8 pad (80B rows, 16B multiple)
    __shared__ __align__(16) bf16 As[2][128 * LDT];
    __shared__ __align__(16) bf16 Bs[2][128 * LDT];

    const int tid  = threadIdx.x;
    const int lane = tid & 31;
    const int w    = tid >> 5;
    const int half = lane >> 4;
    const int lm   = lane & 15;

    const int m0 = blockIdx.y * 128;
    const int n0 = blockIdx.x * 128;
    const int wm = (w >> 2) * 64;   // 0 / 64
    const int wn = (w & 3) * 32;    // 0 / 32 / 64 / 96

    const int ldr  = tid >> 1;           // 0..127 : tile row
    const int lseg = (tid & 1) * 16;     // 0 / 16 : halves

    v8f acc[4][2];
#pragma unroll
    for (int i = 0; i < 4; ++i)
#pragma unroll
        for (int j = 0; j < 2; ++j)
#pragma unroll
            for (int r = 0; r < 8; ++r) acc[i][j][r] = 0.0f;

    // Tile fill: each thread moves 2x16B of A and 2x16B of B.
    auto issue_tile = [&](int k0, int buf) {
        const bf16* ga = A + (size_t)(m0 + ldr) * K + k0 + lseg;
        const bf16* gb = B + (size_t)(n0 + ldr) * K + k0 + lseg;
#if HAVE_ASYNC_LDS
        async16(ga,     &As[buf][ldr * LDT + lseg]);
        async16(ga + 8, &As[buf][ldr * LDT + lseg + 8]);
        async16(gb,     &Bs[buf][ldr * LDT + lseg]);
        async16(gb + 8, &Bs[buf][ldr * LDT + lseg + 8]);
#else
        uint4 a0 = *reinterpret_cast<const uint4*>(ga);
        uint4 a1 = *reinterpret_cast<const uint4*>(ga + 8);
        uint4 b0 = *reinterpret_cast<const uint4*>(gb);
        uint4 b1 = *reinterpret_cast<const uint4*>(gb + 8);
        *reinterpret_cast<uint4*>(&As[buf][ldr * LDT + lseg])     = a0;
        *reinterpret_cast<uint4*>(&As[buf][ldr * LDT + lseg + 8]) = a1;
        *reinterpret_cast<uint4*>(&Bs[buf][ldr * LDT + lseg])     = b0;
        *reinterpret_cast<uint4*>(&Bs[buf][ldr * LDT + lseg + 8]) = b1;
#endif
    };

    issue_tile(0, 0);
    wait_async();
    __syncthreads();

    int buf = 0;
    for (int k0 = 0; k0 < K; k0 += 32) {
        if (k0 + 32 < K) issue_tile(k0 + 32, buf ^ 1);   // overlap next fill

        v16bf afr[4], bfr[2];
#pragma unroll
        for (int i = 0; i < 4; ++i) {
            const int row = wm + i * 16 + lm;
            afr[i] = load16(&As[buf][row * LDT + half * 8],
                            &As[buf][row * LDT + half * 8 + 16]);
        }
#pragma unroll
        for (int j = 0; j < 2; ++j) {
            const int col = wn + j * 16 + lm;
            bfr[j] = load16(&Bs[buf][col * LDT + half * 8],
                            &Bs[buf][col * LDT + half * 8 + 16]);
        }
#pragma unroll
        for (int i = 0; i < 4; ++i)
#pragma unroll
            for (int j = 0; j < 2; ++j)
                acc[i][j] = WMMA_BF16(afr[i], bfr[j], acc[i][j]);

        wait_async();
        __syncthreads();
        buf ^= 1;
    }

#pragma unroll
    for (int i = 0; i < 4; ++i)
#pragma unroll
        for (int j = 0; j < 2; ++j)
#pragma unroll
            for (int r = 0; r < 8; ++r) {
                const int m = m0 + wm + i * 16 + half * 8 + r;
                const int n = n0 + wn + j * 16 + lm;
                C[(size_t)m * N + n] = acc[i][j][r];
            }
}

// ---------------------------------------------------------------------------
// 3) RoPE + layout scatter.
//    qkv f32 [B*S, 3072] -> Qr bf16 [B,NH,S,64] (x 0.125 score scale folded),
//    Kr bf16 [B,NKV,S,64], Vt bf16 [B,NKV,64,S]
// ---------------------------------------------------------------------------
__global__ void rope_scatter(const float* __restrict__ qkv,
                             const float* __restrict__ fr,
                             bf16* __restrict__ Qr, bf16* __restrict__ Kr,
                             bf16* __restrict__ Vt) {
    const int t = blockIdx.x * blockDim.x + threadIdx.x;
    const int total = B_ * S_ * (TK_ / 2);
    if (t >= total) return;
    const int e2 = t % (TK_ / 2);
    const int bs = t / (TK_ / 2);
    const int b = bs / S_, s = bs % S_;
    const int e = e2 * 2;
    const float* q = qkv + (size_t)bs * TK_;

    if (e < D_) {                                   // query section (+RoPE, x0.125)
        const int h = e >> 6, d = e & 63, j = d >> 1;
        const float f0 = fr[(s * 32 + j) * 2 + 0];
        const float f1 = fr[(s * 32 + j) * 2 + 1];
        const float x0 = q[e], x1 = q[e + 1];
        bf16* dst = Qr + (((size_t)(b * NH_ + h) * S_ + s) * HD_ + d);
        dst[0] = (bf16)((x0 * f0 - x1 * f1) * 0.125f);
        dst[1] = (bf16)((x1 * f0 + x0 * f1) * 0.125f);
    } else if (e < D_ + NKV_ * HD_) {               // key section (+RoPE)
        const int ek = e - D_;
        const int h = ek >> 6, d = ek & 63, j = d >> 1;
        const float f0 = fr[(s * 32 + j) * 2 + 0];
        const float f1 = fr[(s * 32 + j) * 2 + 1];
        const float x0 = q[e], x1 = q[e + 1];
        bf16* dst = Kr + (((size_t)(b * NKV_ + h) * S_ + s) * HD_ + d);
        dst[0] = (bf16)(x0 * f0 - x1 * f1);
        dst[1] = (bf16)(x1 * f0 + x0 * f1);
    } else {                                        // value section (transposed)
        const int ev = e - D_ - NKV_ * HD_;
        const int h = ev >> 6, d = ev & 63;
        bf16* base = Vt + ((size_t)(b * NKV_ + h) * HD_) * S_;
        base[(size_t)(d + 0) * S_ + s] = (bf16)q[e + 0];
        base[(size_t)(d + 1) * S_ + s] = (bf16)q[e + 1];
    }
}

// ---------------------------------------------------------------------------
// 4) Causal GQA flash attention.
//    grid = (S/128, B*NH), 256 threads = 8 waves; wave owns 16 query rows.
//    Double-buffered K/V tiles with async fill; P transposed through LDS.
//    Output AO bf16 [B, S, NH, 64]  (== [B*S, D] row-major for the out-proj)
// ---------------------------------------------------------------------------
__global__ __launch_bounds__(256)
void attn_fwd(const bf16* __restrict__ Q, const bf16* __restrict__ Kc,
              const bf16* __restrict__ Vt, bf16* __restrict__ AO) {
    constexpr int LKT = 72;  // 64 hd + 8 pad
    constexpr int LVT = 40;  // 32 keys + 8 pad
    constexpr int LPT = 40;  // 32 keys + 8 pad
    __shared__ __align__(16) bf16 Kl[2][32 * LKT];
    __shared__ __align__(16) bf16 Vl[2][64 * LVT];
    __shared__ __align__(16) bf16 Pl[8][16 * LPT];  // per-wave P staging

    const int tid  = threadIdx.x;
    const int lane = tid & 31;
    const int w    = tid >> 5;
    const int half = lane >> 4;
    const int lm   = lane & 15;

    const int qt  = blockIdx.x;
    const int bh  = blockIdx.y;       // b*NH + h
    const int b   = bh >> 5, h = bh & 31;
    const int kvh = h >> 2;           // GQA: 4 query heads per kv head

    const bf16* Qp = Q  + ((size_t)bh * S_) * HD_;
    const bf16* Kp = Kc + ((size_t)(b * NKV_ + kvh) * S_) * HD_;
    const bf16* Vp = Vt + ((size_t)(b * NKV_ + kvh) * HD_) * S_;

    const int m0 = qt * 128 + w * 16;  // wave's first query row

    const int kr = tid >> 3, ks = (tid & 7) * 8;   // K tile: 32 rows x 64 halves
    const int vr = tid >> 2, vs = (tid & 3) * 8;   // V tile: 64 rows x 32 halves

    auto issue_kv = [&](int k0, int buf) {
#if HAVE_ASYNC_LDS
        async16(&Kp[(size_t)(k0 + kr) * HD_ + ks], &Kl[buf][kr * LKT + ks]);
        async16(&Vp[(size_t)vr * S_ + k0 + vs],    &Vl[buf][vr * LVT + vs]);
#else
        uint4 kk = *reinterpret_cast<const uint4*>(&Kp[(size_t)(k0 + kr) * HD_ + ks]);
        uint4 vv = *reinterpret_cast<const uint4*>(&Vp[(size_t)vr * S_ + k0 + vs]);
        *reinterpret_cast<uint4*>(&Kl[buf][kr * LKT + ks]) = kk;
        *reinterpret_cast<uint4*>(&Vl[buf][vr * LVT + vs]) = vv;
#endif
    };

    // Q A-fragments for hd chunks [0,32) and [32,64), loaded once.
    v16bf qa[2];
#pragma unroll
    for (int c = 0; c < 2; ++c) {
        const bf16* qr = Qp + (size_t)(m0 + lm) * HD_ + c * 32 + half * 8;
        qa[c] = load16(qr, qr + 16);
    }

    float rowmax[8], rowsum[8];
    v8f o[4];
#pragma unroll
    for (int r = 0; r < 8; ++r) { rowmax[r] = -1e30f; rowsum[r] = 0.0f; }
#pragma unroll
    for (int j = 0; j < 4; ++j)
#pragma unroll
        for (int r = 0; r < 8; ++r) o[j][r] = 0.0f;

    const int nkt = qt * 4 + 4;       // causal clip: keys 0 .. qt*128+127
    issue_kv(0, 0);
    wait_async();
    __syncthreads();

    int buf = 0;
    for (int kt = 0; kt < nkt; ++kt) {
        const int k0 = kt * 32;
        if (kt + 1 < nkt) issue_kv(k0 + 32, buf ^ 1);   // overlap next K/V fill

        if (k0 <= m0 + 15) {   // wave-uniform: this key tile intersects causal region
            // S = Q * K^T  (two 16-key subtiles, hd reduced in 2 chunks of 32)
            v8f sfr[2];
#pragma unroll
            for (int j = 0; j < 2; ++j)
#pragma unroll
                for (int r = 0; r < 8; ++r) sfr[j][r] = 0.0f;
#pragma unroll
            for (int j = 0; j < 2; ++j)
#pragma unroll
                for (int c = 0; c < 2; ++c) {
                    const bf16* kp2 = &Kl[buf][(j * 16 + lm) * LKT + c * 32 + half * 8];
                    v16bf kb = load16(kp2, kp2 + 16);
                    sfr[j] = WMMA_BF16(qa[c], kb, sfr[j]);
                }

            // online softmax over this 32-key chunk (rows live in 16-lane halves)
#pragma unroll
            for (int r = 0; r < 8; ++r) {
                const int qrow = m0 + half * 8 + r;
                float v0 = sfr[0][r]; if (k0 + lm      > qrow) v0 = -1e30f;
                float v1 = sfr[1][r]; if (k0 + 16 + lm > qrow) v1 = -1e30f;
                float mx = fmaxf(v0, v1);
#pragma unroll
                for (int d = 1; d < 16; d <<= 1) mx = fmaxf(mx, __shfl_xor(mx, d, 32));
                const float nm   = fmaxf(rowmax[r], mx);
                const float corr = __expf(rowmax[r] - nm);
                const float p0 = __expf(v0 - nm);
                const float p1 = __expf(v1 - nm);
                float ps = p0 + p1;
#pragma unroll
                for (int d = 1; d < 16; d <<= 1) ps += __shfl_xor(ps, d, 32);
                rowsum[r] = rowsum[r] * corr + ps;
                rowmax[r] = nm;
#pragma unroll
                for (int j = 0; j < 4; ++j) o[j][r] *= corr;
                // stage P (C-layout -> LDS row-major [q][key]) for A-fragment reload
                Pl[w][(half * 8 + r) * LPT + lm]      = (bf16)p0;
                Pl[w][(half * 8 + r) * LPT + 16 + lm] = (bf16)p1;
            }

            // O += P * V   (A from wave-private LDS, B from Vt tile)
            const bf16* pr = &Pl[w][lm * LPT + half * 8];
            v16bf pf = load16(pr, pr + 16);
#pragma unroll
            for (int j = 0; j < 4; ++j) {
                const bf16* vp2 = &Vl[buf][(j * 16 + lm) * LVT + half * 8];
                v16bf vb = load16(vp2, vp2 + 16);
                o[j] = WMMA_BF16(pf, vb, o[j]);
            }
        }

        wait_async();
        __syncthreads();
        buf ^= 1;
    }

    // normalize + store AO[b][s][h][hd]
#pragma unroll
    for (int r = 0; r < 8; ++r) {
        const float inv = 1.0f / rowsum[r];
        const int m = m0 + half * 8 + r;
        bf16* dst = AO + ((size_t)(b * S_ + m) * NH_ + h) * HD_;
#pragma unroll
        for (int j = 0; j < 4; ++j)
            dst[j * 16 + lm] = (bf16)(o[j][r] * inv);
    }
}

// ---------------------------------------------------------------------------
// Launch
// ---------------------------------------------------------------------------
extern "C" void kernel_launch(void* const* d_in, const int* in_sizes, int n_in,
                              void* d_out, int out_size, void* d_ws, size_t ws_size,
                              hipStream_t stream) {
    (void)in_sizes; (void)n_in; (void)out_size; (void)ws_size;
    const float* x    = (const float*)d_in[0];
    const float* fr   = (const float*)d_in[1];
    const float* wqkv = (const float*)d_in[2];
    const float* wo   = (const float*)d_in[3];
    float* out = (float*)d_out;

    const size_t MSZ = (size_t)B_ * S_;  // 4096 rows

    char* ws = (char*)d_ws;
    size_t off = 0;
    auto take = [&](size_t bytes) -> char* {
        char* p = ws + off;
        off += (bytes + 255) & ~(size_t)255;
        return p;
    };
    bf16*  xb  = (bf16*) take(MSZ * D_ * sizeof(bf16));
    bf16*  wqb = (bf16*) take((size_t)TK_ * D_ * sizeof(bf16));
    bf16*  wob = (bf16*) take((size_t)D_ * D_ * sizeof(bf16));
    float* qkv = (float*)take(MSZ * TK_ * sizeof(float));
    bf16*  Qr  = (bf16*) take(MSZ * D_ * sizeof(bf16));
    bf16*  Kr  = (bf16*) take(MSZ * (size_t)(NKV_ * HD_) * sizeof(bf16));
    bf16*  Vt  = (bf16*) take(MSZ * (size_t)(NKV_ * HD_) * sizeof(bf16));
    bf16*  AO  = (bf16*) take(MSZ * D_ * sizeof(bf16));

    const int n1 = B_ * S_ * D_;
    cast_f32_bf16<<<(n1 + 255) / 256, 256, 0, stream>>>(x, xb, n1);
    const int n2 = TK_ * D_;
    cast_f32_bf16<<<(n2 + 255) / 256, 256, 0, stream>>>(wqkv, wqb, n2);
    const int n3 = D_ * D_;
    cast_f32_bf16<<<(n3 + 255) / 256, 256, 0, stream>>>(wo, wob, n3);

    // qkv = x @ wqkv^T : [4096,2048] x [3072,2048]^T
    gemm_bf16_nt<<<dim3(TK_ / 128, (int)(MSZ / 128)), 256, 0, stream>>>(
        xb, wqb, qkv, (int)MSZ, TK_, D_);

    // RoPE + scatter into attention layouts
    const int nr = (int)(MSZ * (TK_ / 2));
    rope_scatter<<<(nr + 255) / 256, 256, 0, stream>>>(qkv, fr, Qr, Kr, Vt);

    // flash attention
    attn_fwd<<<dim3(S_ / 128, B_ * NH_), 256, 0, stream>>>(Qr, Kr, Vt, AO);

    // out = AO @ wo^T : [4096,2048] x [2048,2048]^T
    gemm_bf16_nt<<<dim3(D_ / 128, (int)(MSZ / 128)), 256, 0, stream>>>(
        AO, wob, out, (int)MSZ, D_, D_);
}
